// Network_41927470743685
// MI455X (gfx1250) — matrix operand order repaired and checked by reading, not compiled, not run on hardware
//
#include <hip/hip_runtime.h>
#include <hip/hip_bf16.h>

#define NB   4
#define NPT  16384
#define KNB  16
#define NPTS (NB * NPT)

typedef __attribute__((ext_vector_type(16))) _Float16 v16h;
typedef __attribute__((ext_vector_type(8)))  float    v8f;

static __device__ inline v8f wmma_f16(v16h a, v16h b, v8f c) {
  // D = A(16x32,f16) * B(32x16,f16) + C(16x16,f32)
  return __builtin_amdgcn_wmma_f32_16x16x32_f16(false, a, false, b, (short)0, c,
                                                false, false);
}

// B fragment (16-bit B-matrix 32x16 layout): lane = half*16 + n,
// VGPR v holds K = (v/4)*16 + half*8 + (v%4)*2 (+1).  B[k][n] = W[(n+n0)*row_len + k_base+k]
static __device__ inline v16h build_bfrag(const float* __restrict__ W,
                                          int row_len, int n0, int k_base) {
  int lane = threadIdx.x & 31;
  int nn = (lane & 15) + n0;
  int kh = (lane >> 4) & 1;
  v16h b;
#pragma unroll
  for (int v = 0; v < 8; ++v) {
    int k0 = k_base + (v >> 2) * 16 + kh * 8 + (v & 3) * 2;
    float f0 = (k0     - k_base < row_len - k_base && k0     < row_len) ? W[nn * row_len + k0    ] : 0.f;
    float f1 = (k0 + 1 < row_len) ? W[nn * row_len + k0 + 1] : 0.f;
    b[2 * v]     = (_Float16)f0;
    b[2 * v + 1] = (_Float16)f1;
  }
  return b;
}

// A fragment (16-bit A-matrix 16x32 layout): lane m = L&15, half = L>>4,
// VGPR v: K = (v/4)*16 + half*8 + (v%4)*2 (+1).  S is row-major [16][stride] f16.
static __device__ inline v16h build_afrag(const _Float16* S, int stride, int k_base) {
  int lane = threadIdx.x & 31;
  int m = lane & 15;
  int half = (lane >> 4) & 1;
  v16h a;
#pragma unroll
  for (int v = 0; v < 8; ++v) {
    int k0 = k_base + (v >> 2) * 16 + half * 8 + (v & 3) * 2;
    a[2 * v]     = S[m * stride + k0];
    a[2 * v + 1] = S[m * stride + k0 + 1];
  }
  return a;
}

// ---------------- Kernel 1: f_pc = relu(bn(mlp1 @ feature)) -> f16 [b][n][32]
__global__ __launch_bounds__(256) void k_mlp1(const float* __restrict__ feature,
                                              const float* __restrict__ W,
                                              const float* __restrict__ s,
                                              const float* __restrict__ bia,
                                              _Float16* __restrict__ f_pc) {
  int lane = threadIdx.x & 31;
  int wv = threadIdx.x >> 5;
  int tile = blockIdx.x * 8 + wv;     // 16 points per wave
  int p0 = tile * 16;
  int b = p0 >> 14;
  int n0 = p0 & (NPT - 1);

  v16h bw0 = build_bfrag(W, 32, 0, 0);
  v16h bw1 = build_bfrag(W, 32, 16, 0);

  int m = lane & 15, half = lane >> 4;
  v16h a;
#pragma unroll
  for (int v = 0; v < 8; ++v) {
    int k0 = (v >> 2) * 16 + half * 8 + (v & 3) * 2;
    a[2 * v]     = (_Float16)feature[((size_t)b * 32 + k0)     * NPT + n0 + m];
    a[2 * v + 1] = (_Float16)feature[((size_t)b * 32 + k0 + 1) * NPT + n0 + m];
  }
  v8f z = {};
  v8f d0 = wmma_f16(a, bw0, z);
  v8f d1 = wmma_f16(a, bw1, z);

  int nI = lane & 15, mb = (lane >> 4) * 8;
#pragma unroll
  for (int t = 0; t < 2; ++t) {
    v8f d = t ? d1 : d0;
    int o = nI + 16 * t;
    float sc = s[o], bi = bia[o];
#pragma unroll
    for (int v = 0; v < 8; ++v) {
      float y = fmaxf(d[v] * sc + bi, 0.f);
      f_pc[((size_t)b * NPT + n0 + mb + v) * 32 + o] = (_Float16)y;
    }
  }
}

// relative position encoding: lanes 0..15 fill aRow[j][0..31] (10 real + pad).
// every lane returns the neighbor index for j = lane&15.
static __device__ inline int relpos(const float* __restrict__ xyz,
                                    const int* __restrict__ nidx, int b, int n,
                                    _Float16* aRow) {
  int lane = threadIdx.x & 31;
  int j = lane & 15;
  int idx = nidx[((size_t)b * NPT + n) * KNB + j];
  if (lane < 16) {
    const float* pc = xyz + ((size_t)b * NPT + n) * 3;
    const float* pn = xyz + ((size_t)b * NPT + idx) * 3;
    float cx = pc[0], cy = pc[1], cz = pc[2];
    float nx = pn[0], ny = pn[1], nz = pn[2];
    float rx = cx - nx, ry = cy - ny, rz = cz - nz;
    float rd = sqrtf(rx * rx + ry * ry + rz * rz);
    _Float16* row = aRow + j * 32;
    row[0] = (_Float16)rd; row[1] = (_Float16)rx; row[2] = (_Float16)ry;
    row[3] = (_Float16)rz; row[4] = (_Float16)cx; row[5] = (_Float16)cy;
    row[6] = (_Float16)cz; row[7] = (_Float16)nx; row[8] = (_Float16)ny;
    row[9] = (_Float16)nz;
#pragma unroll
    for (int q = 10; q < 32; ++q) row[q] = (_Float16)0.f;
  }
  return idx;
}

// attention pool core: 8 WMMAs (fc), softmax over neighbors, weighted agg -> aggS[64]
static __device__ inline void att_core(const v16h b_fc[4][2], const _Float16* fcRow,
                                       float* aggS) {
  int lane = threadIdx.x & 31;
  int nI = lane & 15, mb = (lane >> 4) * 8;
  v8f z = {};
  v16h a0 = build_afrag(fcRow, 64, 0);
  v16h a1 = build_afrag(fcRow, 64, 32);
#pragma unroll
  for (int t = 0; t < 4; ++t) {
    v8f d = wmma_f16(a0, b_fc[t][0], z);
    d = wmma_f16(a1, b_fc[t][1], d);
    float mx = d[0];
#pragma unroll
    for (int v = 1; v < 8; ++v) mx = fmaxf(mx, d[v]);
    mx = fmaxf(mx, __shfl_xor(mx, 16, 32));
    float e[8], ssum = 0.f;
#pragma unroll
    for (int v = 0; v < 8; ++v) { e[v] = __expf(d[v] - mx); ssum += e[v]; }
    ssum += __shfl_xor(ssum, 16, 32);
    float inv = 1.f / ssum;
    int c = nI + 16 * t;
    float pa = 0.f;
#pragma unroll
    for (int v = 0; v < 8; ++v) pa += (float)fcRow[(mb + v) * 64 + c] * (e[v] * inv);
    pa += __shfl_xor(pa, 16, 32);
    if (lane < 16) aggS[c] = pa;
  }
}

// ---------------- Kernel 2: bb1 + att_pool1 -> f_agg f16 [b][n][32]
__global__ __launch_bounds__(256) void k_stage1(
    const float* __restrict__ xyz, const int* __restrict__ nidx,
    const float* __restrict__ bb1_W, const float* __restrict__ bb1_s,
    const float* __restrict__ bb1_b, const float* __restrict__ fc_W,
    const float* __restrict__ att_W, const float* __restrict__ att_s,
    const float* __restrict__ att_b, const _Float16* __restrict__ f_pc,
    _Float16* __restrict__ f_agg) {
  __shared__ __align__(16) _Float16 aS[8][16][32];
  __shared__ __align__(16) _Float16 fcS[8][16][64];
  __shared__ float aggS[8][64];

  int lane = threadIdx.x & 31;
  int wv = threadIdx.x >> 5;
  int p = blockIdx.x * 8 + wv;
  int b = p >> 14, n = p & (NPT - 1);

  v16h b_bb1[2];
  b_bb1[0] = build_bfrag(bb1_W, 10, 0, 0);
  b_bb1[1] = build_bfrag(bb1_W, 10, 16, 0);
  v16h b_fc[4][2];
#pragma unroll
  for (int t = 0; t < 4; ++t) {
    b_fc[t][0] = build_bfrag(fc_W, 64, 16 * t, 0);
    b_fc[t][1] = build_bfrag(fc_W, 64, 16 * t, 32);
  }

  int idx = relpos(xyz, nidx, b, n, &aS[wv][0][0]);
  {  // gather f_pc (32 f16 per neighbor) -> fcS channels 0..31
    int j = lane & 15, hi = lane >> 4;
    const uint4* src = (const uint4*)(f_pc + ((size_t)b * NPT + idx) * 32 + hi * 16);
    uint4 q0 = src[0], q1 = src[1];
    uint4* dst = (uint4*)&fcS[wv][j][hi * 16];
    dst[0] = q0; dst[1] = q1;
  }
  __syncthreads();

  // bb1: f_xyz = relu(bn(A @ bb1_W^T)) -> fcS channels 32..63
  v16h a = build_afrag(&aS[wv][0][0], 32, 0);
  v8f z = {};
  int nI = lane & 15, mb = (lane >> 4) * 8;
#pragma unroll
  for (int t = 0; t < 2; ++t) {
    v8f d = wmma_f16(a, b_bb1[t], z);
    int o = nI + 16 * t;
    float sc = bb1_s[o], bi = bb1_b[o];
#pragma unroll
    for (int v = 0; v < 8; ++v) {
      float y = fmaxf(d[v] * sc + bi, 0.f);
      fcS[wv][mb + v][32 + o] = (_Float16)y;
    }
  }
  __syncthreads();

  att_core(b_fc, &fcS[wv][0][0], &aggS[wv][0]);
  __syncthreads();

  // att1 conv-bn-relu matvec: one output channel per lane
  float acc = 0.f;
#pragma unroll
  for (int c = 0; c < 64; ++c) acc += aggS[wv][c] * att_W[lane * 64 + c];
  float y = fmaxf(acc * att_s[lane] + att_b[lane], 0.f);
  f_agg[((size_t)b * NPT + n) * 32 + lane] = (_Float16)y;
}

// ---------------- Kernel 3: recompute f_xyz, bb2, att_pool2, mlp2 + shortcut
__global__ __launch_bounds__(256) void k_stage2(
    const float* __restrict__ feature, const float* __restrict__ xyz,
    const int* __restrict__ nidx, const float* __restrict__ bb1_W,
    const float* __restrict__ bb1_s, const float* __restrict__ bb1_b,
    const float* __restrict__ bb2_W, const float* __restrict__ bb2_s,
    const float* __restrict__ bb2_b, const float* __restrict__ fc_W,
    const float* __restrict__ att_W, const float* __restrict__ att_s,
    const float* __restrict__ att_b, const float* __restrict__ mlp2_W,
    const float* __restrict__ mlp2_s, const float* __restrict__ mlp2_b,
    const float* __restrict__ sc_W, const float* __restrict__ sc_s,
    const float* __restrict__ sc_b, const _Float16* __restrict__ f_agg,
    float* __restrict__ out) {
  __shared__ __align__(16) _Float16 aS[8][16][32];
  __shared__ __align__(16) _Float16 fcS[8][16][64];
  __shared__ float aggS[8][64];
  __shared__ float lfaS[8][64];
  __shared__ float featS[8][32];

  int lane = threadIdx.x & 31;
  int wv = threadIdx.x >> 5;
  int p = blockIdx.x * 8 + wv;
  int b = p >> 14, n = p & (NPT - 1);

  v16h b_bb1[2], b_bb2[2];
  b_bb1[0] = build_bfrag(bb1_W, 10, 0, 0);
  b_bb1[1] = build_bfrag(bb1_W, 10, 16, 0);
  b_bb2[0] = build_bfrag(bb2_W, 32, 0, 0);
  b_bb2[1] = build_bfrag(bb2_W, 32, 16, 0);
  v16h b_fc[4][2];
#pragma unroll
  for (int t = 0; t < 4; ++t) {
    b_fc[t][0] = build_bfrag(fc_W, 64, 16 * t, 0);
    b_fc[t][1] = build_bfrag(fc_W, 64, 16 * t, 32);
  }

  int idx = relpos(xyz, nidx, b, n, &aS[wv][0][0]);
  {  // gather f_agg -> fcS channels 0..31
    int j = lane & 15, hi = lane >> 4;
    const uint4* src = (const uint4*)(f_agg + ((size_t)b * NPT + idx) * 32 + hi * 16);
    uint4 q0 = src[0], q1 = src[1];
    uint4* dst = (uint4*)&fcS[wv][j][hi * 16];
    dst[0] = q0; dst[1] = q1;
  }
  featS[wv][lane] = feature[((size_t)b * 32 + lane) * NPT + n];
  __syncthreads();

  v8f z = {};
  int nI = lane & 15, mb = (lane >> 4) * 8;

  // bb1 -> f_xyz (overwrite aS rows with f16 result)
  v16h a = build_afrag(&aS[wv][0][0], 32, 0);
#pragma unroll
  for (int t = 0; t < 2; ++t) {
    v8f d = wmma_f16(a, b_bb1[t], z);
    int o = nI + 16 * t;
    float sc = bb1_s[o], bi = bb1_b[o];
#pragma unroll
    for (int v = 0; v < 8; ++v) {
      float y = fmaxf(d[v] * sc + bi, 0.f);
      aS[wv][mb + v][o] = (_Float16)y;
    }
  }
  __syncthreads();

  // bb2 -> f_xyz2 -> fcS channels 32..63
  v16h a2 = build_afrag(&aS[wv][0][0], 32, 0);
#pragma unroll
  for (int t = 0; t < 2; ++t) {
    v8f d = wmma_f16(a2, b_bb2[t], z);
    int o = nI + 16 * t;
    float sc = bb2_s[o], bi = bb2_b[o];
#pragma unroll
    for (int v = 0; v < 8; ++v) {
      float y = fmaxf(d[v] * sc + bi, 0.f);
      fcS[wv][mb + v][32 + o] = (_Float16)y;
    }
  }
  __syncthreads();

  att_core(b_fc, &fcS[wv][0][0], &aggS[wv][0]);
  __syncthreads();

  // att2 conv-bn-relu matvec (64 outputs, 2 per lane) -> lfaS
#pragma unroll
  for (int r = 0; r < 2; ++r) {
    int o = lane + 32 * r;
    float acc = 0.f;
#pragma unroll
    for (int c = 0; c < 64; ++c) acc += aggS[wv][c] * att_W[o * 64 + c];
    lfaS[wv][o] = fmaxf(acc * att_s[o] + att_b[o], 0.f);
  }
  __syncthreads();

  // mlp2 + shortcut + leaky relu (128 outputs, 4 per lane)
#pragma unroll
  for (int r = 0; r < 4; ++r) {
    int o = lane + 32 * r;
    float fo = 0.f;
#pragma unroll
    for (int c = 0; c < 64; ++c) fo += lfaS[wv][c] * mlp2_W[o * 64 + c];
    fo = fo * mlp2_s[o] + mlp2_b[o];
    float sh = 0.f;
#pragma unroll
    for (int c = 0; c < 32; ++c) sh += featS[wv][c] * sc_W[o * 32 + c];
    sh = sh * sc_s[o] + sc_b[o];
    float y = fo + sh;
    out[((size_t)b * 128 + o) * NPT + n] = (y > 0.f) ? y : 0.2f * y;
  }
}

extern "C" void kernel_launch(void* const* d_in, const int* in_sizes, int n_in,
                              void* d_out, int out_size, void* d_ws, size_t ws_size,
                              hipStream_t stream) {
  (void)in_sizes; (void)n_in; (void)out_size; (void)ws_size;
  const float* feature   = (const float*)d_in[0];
  const float* xyz       = (const float*)d_in[1];
  const int*   nidx      = (const int*)d_in[2];
  const float* mlp1_W    = (const float*)d_in[3];
  const float* mlp1_s    = (const float*)d_in[4];
  const float* mlp1_b    = (const float*)d_in[5];
  const float* bb1_W     = (const float*)d_in[6];
  const float* bb1_s     = (const float*)d_in[7];
  const float* bb1_b     = (const float*)d_in[8];
  const float* att1_fc_W = (const float*)d_in[9];
  const float* att1_W    = (const float*)d_in[10];
  const float* att1_s    = (const float*)d_in[11];
  const float* att1_b    = (const float*)d_in[12];
  const float* bb2_W     = (const float*)d_in[13];
  const float* bb2_s     = (const float*)d_in[14];
  const float* bb2_b     = (const float*)d_in[15];
  const float* att2_fc_W = (const float*)d_in[16];
  const float* att2_W    = (const float*)d_in[17];
  const float* att2_s    = (const float*)d_in[18];
  const float* att2_b    = (const float*)d_in[19];
  const float* mlp2_W    = (const float*)d_in[20];
  const float* mlp2_s    = (const float*)d_in[21];
  const float* mlp2_b    = (const float*)d_in[22];
  const float* sc_W      = (const float*)d_in[23];
  const float* sc_s      = (const float*)d_in[24];
  const float* sc_b      = (const float*)d_in[25];

  _Float16* f_pc  = (_Float16*)d_ws;                 // [B*N][32] f16 = 4 MB
  _Float16* f_agg = f_pc + (size_t)NPTS * 32;        // [B*N][32] f16 = 4 MB

  k_mlp1<<<NPTS / 16 / 8, 256, 0, stream>>>(feature, mlp1_W, mlp1_s, mlp1_b, f_pc);
  k_stage1<<<NPTS / 8, 256, 0, stream>>>(xyz, nidx, bb1_W, bb1_s, bb1_b,
                                         att1_fc_W, att1_W, att1_s, att1_b,
                                         f_pc, f_agg);
  k_stage2<<<NPTS / 8, 256, 0, stream>>>(feature, xyz, nidx, bb1_W, bb1_s, bb1_b,
                                         bb2_W, bb2_s, bb2_b, att2_fc_W, att2_W,
                                         att2_s, att2_b, mlp2_W, mlp2_s, mlp2_b,
                                         sc_W, sc_s, sc_b, f_agg, (float*)d_out);
}
